// SparseMemoryAttention_28174985462331
// MI455X (gfx1250) — matrix operand order repaired and checked by reading, not compiled, not run on hardware
//
#include <hip/hip_runtime.h>
#include <hip/hip_bf16.h>
#include <stddef.h>
#include <stdint.h>

// ---------------------------------------------------------------------------
// SparseMemoryAttention for MI455X (gfx1250, wave32, WMMA).
//
// B=2, T=1024, D=2048, H=16, dh=128, N=2048, WINDOW=256, TOPK=8.
// ~195 GFLOP total, ~170 GFLOP in 7 NT GEMMs -> v_wmma_f32_16x16x32_f16
// (f16 in, f32 accumulate).  Working set is L2-resident (192MB) so the matrix
// pipes are the limiter; GEMM tiles are staged LDS-side with the CDNA5 async
// copy path (GLOBAL_LOAD_ASYNC_TO_LDS_B128 + s_wait_asynccnt), double-buffered
// so the next K-tile is in flight while the current one feeds the WMMAs.
// A sched_barrier splits LDS fragment loads from the WMMA chain so the 8
// WMMAs issue back-to-back.  Irregular attention (banded softmax, top-8,
// gather) runs on VALU+LDS.  Workspace requirement: ~184 MB in d_ws.
// ---------------------------------------------------------------------------

#ifndef SMA_USE_ASYNC
#define SMA_USE_ASYNC 1   // 1: global_load_async_to_lds_b128; 0: sync copy
#endif

typedef __attribute__((ext_vector_type(16))) _Float16 v16h;
typedef __attribute__((ext_vector_type(8)))  _Float16 v8h;
typedef __attribute__((ext_vector_type(8)))  float    v8f;

namespace sma {
constexpr int B_   = 2;
constexpr int T_   = 1024;
constexpr int D_   = 2048;
constexpr int H_   = 16;
constexpr int DH   = 128;          // head dim
constexpr int NMEM = 2048;
constexpr int WIN  = 256;
constexpr int TOPK = 8;
constexpr float SCALE = 0.08838834764831845f;   // 1/sqrt(128)

constexpr size_t BT  = (size_t)B_ * T_;          // 2048
constexpr size_t BN  = (size_t)B_ * NMEM;        // 4096
constexpr size_t BTD = BT * D_;                  // 4.19M
constexpr size_t BND = BN * D_;                  // 8.39M
constexpr size_t DD  = (size_t)D_ * D_;          // 4.19M
constexpr size_t D2D = (size_t)D_ * 2 * D_;      // 8.39M
}

__device__ __forceinline__ void sma_sched_fence() {
#if __has_builtin(__builtin_amdgcn_sched_barrier)
  __builtin_amdgcn_sched_barrier(0);
#endif
}

// --------------------------- f32 -> f16 convert -----------------------------
__global__ void sma_cvt_f16(const float* __restrict__ src,
                            _Float16* __restrict__ dst, size_t n) {
  size_t i = (size_t)blockIdx.x * blockDim.x + threadIdx.x;
  size_t stride = (size_t)gridDim.x * blockDim.x;
  for (; i < n; i += stride) dst[i] = (_Float16)src[i];
}

// --------------------------- RoPE + f32 -> f16 ------------------------------
__global__ void sma_rope_f16(const float* __restrict__ X,
                             const float* __restrict__ cosb,
                             const float* __restrict__ sinb,
                             _Float16* __restrict__ out) {
  using namespace sma;
  size_t i = (size_t)blockIdx.x * blockDim.x + threadIdx.x;
  size_t stride = (size_t)gridDim.x * blockDim.x;
  for (; i < BTD; i += stride) {
    int col = (int)(i % D_);
    size_t bt = i / D_;
    int d = col & (DH - 1);
    float c = cosb[bt * DH + d];
    float s = sinb[bt * DH + d];
    float x = X[i];
    float xr = (d < DH / 2) ? -X[i + DH / 2] : X[i - DH / 2];
    out[i] = (_Float16)(x * c + xr * s);
  }
}

// ------------------------------ WMMA NT GEMM --------------------------------
// C[M,N] = A[M,K] @ W[N,K]^T (+bias).  M,N multiples of 128, K multiple of 32.
// grid = (N/128, M/128), 256 threads (8 waves).  Wave w owns rows
// [m0+16w, m0+16w+16) x all 128 cols.  Double-buffered LDS K-tiles filled by
// async-to-LDS DMA; 8 B-fragments preloaded, sched_barrier keeps the WMMAs
// issuing back-to-back behind a single DS wait.
__global__ __launch_bounds__(256) void sma_gemm_nt(
    const _Float16* __restrict__ A, const _Float16* __restrict__ W,
    float* __restrict__ Cf, _Float16* __restrict__ Ch,
    const float* __restrict__ bias, int K, int ldc) {
  constexpr int LDA = 40;  // padded stride (halfs) to spread LDS banks
  __shared__ _Float16 As[2][128][LDA];
  __shared__ _Float16 Bs[2][128][LDA];

  const int tid  = threadIdx.x;
  const int lane = tid & 31;
  const int wv   = tid >> 5;
  const int r    = lane & 15;   // row-in-frag / col-in-frag
  const int half = lane >> 4;   // half-wave select
  const int n0 = blockIdx.x * 128;
  const int m0 = blockIdx.y * 128;

  // Tile staging: 128 rows x 64B per matrix = 512 x 16B chunks; 256 threads
  // -> each thread moves chunk c0=tid and c1=tid+256 for both A and B.
  const int row0 = tid >> 2;                 // 0..63
  const int row1 = 64 + (tid >> 2);          // 64..127
  const int ch   = (tid & 3) * 8;            // chunk offset within row (f16)

  v8f acc[8] = {};

#if SMA_USE_ASYNC
  auto issue_tile = [&](int buf, int k0) {
    uint32_t lA0 = (uint32_t)(uintptr_t)&As[buf][row0][ch];
    uint32_t lA1 = (uint32_t)(uintptr_t)&As[buf][row1][ch];
    uint32_t lB0 = (uint32_t)(uintptr_t)&Bs[buf][row0][ch];
    uint32_t lB1 = (uint32_t)(uintptr_t)&Bs[buf][row1][ch];
    uint32_t gA0 = (uint32_t)(((size_t)(m0 + row0) * K + k0 + ch) * 2);
    uint32_t gA1 = (uint32_t)(((size_t)(m0 + row1) * K + k0 + ch) * 2);
    uint32_t gB0 = (uint32_t)(((size_t)(n0 + row0) * K + k0 + ch) * 2);
    uint32_t gB1 = (uint32_t)(((size_t)(n0 + row1) * K + k0 + ch) * 2);
    asm volatile("global_load_async_to_lds_b128 %0, %1, %2"
                 :: "v"(lA0), "v"(gA0), "s"(A) : "memory");
    asm volatile("global_load_async_to_lds_b128 %0, %1, %2"
                 :: "v"(lA1), "v"(gA1), "s"(A) : "memory");
    asm volatile("global_load_async_to_lds_b128 %0, %1, %2"
                 :: "v"(lB0), "v"(gB0), "s"(W) : "memory");
    asm volatile("global_load_async_to_lds_b128 %0, %1, %2"
                 :: "v"(lB1), "v"(gB1), "s"(W) : "memory");
  };
#else
  auto issue_tile = [&](int buf, int k0) {
    *(uint4*)&As[buf][row0][ch] = *(const uint4*)&A[(size_t)(m0 + row0) * K + k0 + ch];
    *(uint4*)&As[buf][row1][ch] = *(const uint4*)&A[(size_t)(m0 + row1) * K + k0 + ch];
    *(uint4*)&Bs[buf][row0][ch] = *(const uint4*)&W[(size_t)(n0 + row0) * K + k0 + ch];
    *(uint4*)&Bs[buf][row1][ch] = *(const uint4*)&W[(size_t)(n0 + row1) * K + k0 + ch];
  };
#endif

  const int nkt = K / 32;
  issue_tile(0, 0);

  for (int kt = 0; kt < nkt; ++kt) {
    const int cur = kt & 1;
    if (kt + 1 < nkt) {
      issue_tile(cur ^ 1, (kt + 1) * 32);
#if SMA_USE_ASYNC
      // 4 async ops/wave in flight for the next tile; current tile drained.
      asm volatile("s_wait_asynccnt 0x4" ::: "memory");
#endif
    } else {
#if SMA_USE_ASYNC
      asm volatile("s_wait_asynccnt 0x0" ::: "memory");
#endif
    }
    __syncthreads();

    // A fragment (16x32 f16): lane holds row m0+16w+r; elements 0-15 are
    // K = half*16 .. +15, contiguous.
    const _Float16* ap = &As[cur][wv * 16 + r][half * 16];
    v8h a0 = *(const v8h*)ap;
    v8h a1 = *(const v8h*)(ap + 8);
    v16h afrag = __builtin_shufflevector(a0, a1, 0, 1, 2, 3, 4, 5, 6, 7,
                                         8, 9, 10, 11, 12, 13, 14, 15);

    // Preload all 8 B fragments (32x16 f16 each: lane holds col r of W-rows
    // nb*16+r; elements 0-15 are K = half*16 .. +15, contiguous).
    v16h bfr[8];
#pragma unroll
    for (int nb = 0; nb < 8; ++nb) {
      const _Float16* bp = &Bs[cur][nb * 16 + r][half * 16];
      v8h b0 = *(const v8h*)bp;
      v8h b1 = *(const v8h*)(bp + 8);
      bfr[nb] = __builtin_shufflevector(b0, b1, 0, 1, 2, 3, 4, 5, 6, 7,
                                        8, 9, 10, 11, 12, 13, 14, 15);
    }
    // Keep all ds_loads before, all WMMAs after: one DS wait, then the 8
    // WMMAs (distinct accumulators, no D->A/B hazard) issue back-to-back.
    sma_sched_fence();
#pragma unroll
    for (int nb = 0; nb < 8; ++nb) {
      acc[nb] = __builtin_amdgcn_wmma_f32_16x16x32_f16(
          false, afrag, false, bfr[nb], (short)0, acc[nb], false, false);
    }
    sma_sched_fence();
    __syncthreads();  // buffer 'cur' free for reuse by iteration kt+2's issue
  }

  // C layout: lane holds col = lane%16; VGPR e holds row e + 8*half.
#pragma unroll
  for (int nb = 0; nb < 8; ++nb) {
    int col = n0 + nb * 16 + r;
    float bv = bias ? bias[col] : 0.0f;
#pragma unroll
    for (int e = 0; e < 8; ++e) {
      int row = m0 + wv * 16 + half * 8 + e;
      float v = acc[nb][e] + bv;
      if (Cf) Cf[(size_t)row * ldc + col] = v;
      else    Ch[(size_t)row * ldc + col] = (_Float16)v;
    }
  }
}

// --------------------- local banded attention (window 256) ------------------
__global__ __launch_bounds__(256) void sma_local_attn(
    const _Float16* __restrict__ q16, const _Float16* __restrict__ k16,
    const _Float16* __restrict__ v16, _Float16* __restrict__ olocal) {
  using namespace sma;
  constexpr int SMAX = 272;  // max keys touched by a 16-query tile
  __shared__ float qs[16][DH];
  __shared__ float S[16][SMAX];

  const int tid = threadIdx.x;
  const int qt = blockIdx.x, h = blockIdx.y, b = blockIdx.z;
  const int i0 = qt * 16;
  const int jlo = (i0 - (WIN - 1) > 0) ? i0 - (WIN - 1) : 0;
  const int nk = (i0 + 15) - jlo + 1;

  for (int p = tid; p < 16 * DH; p += 256) {
    int qi = p >> 7, d = p & (DH - 1);
    qs[qi][d] = (float)q16[((size_t)(b * T_ + i0 + qi)) * D_ + h * DH + d];
  }
  __syncthreads();

  for (int p = tid; p < 16 * nk; p += 256) {
    int qi = p / nk, jj = p % nk;
    int i = i0 + qi, j = jlo + jj;
    float val = -1e30f;
    if (j <= i && j > i - WIN) {
      const _Float16* kp = &k16[((size_t)(b * T_ + j)) * D_ + h * DH];
      float sum = 0.0f;
#pragma unroll 8
      for (int d = 0; d < DH; ++d) sum += qs[qi][d] * (float)kp[d];
      val = sum * SCALE;
    }
    S[qi][jj] = val;
  }
  __syncthreads();

  if (tid < 16) {
    float mx = -1e30f;
    for (int jj = 0; jj < nk; ++jj) mx = fmaxf(mx, S[tid][jj]);
    float sum = 0.0f;
    for (int jj = 0; jj < nk; ++jj) {
      float e = __expf(S[tid][jj] - mx);
      S[tid][jj] = e;
      sum += e;
    }
    float inv = 1.0f / sum;
    for (int jj = 0; jj < nk; ++jj) S[tid][jj] *= inv;
  }
  __syncthreads();

  for (int p = tid; p < 16 * DH; p += 256) {
    int qi = p >> 7, d = p & (DH - 1);
    float acc = 0.0f;
    for (int jj = 0; jj < nk; ++jj)
      acc += S[qi][jj] *
             (float)v16[((size_t)(b * T_ + jlo + jj)) * D_ + h * DH + d];
    olocal[((size_t)(b * T_ + i0 + qi)) * D_ + h * DH + d] = (_Float16)acc;
  }
}

// ---------------- memory attention: sim -> top-8 -> softmax -> gather -------
__global__ __launch_bounds__(256) void sma_mem_attn(
    const _Float16* __restrict__ q16, const _Float16* __restrict__ kmem,
    const _Float16* __restrict__ vmem, _Float16* __restrict__ cat) {
  using namespace sma;
  __shared__ float qs[16][DH];
  __shared__ float simc[16][256];
  __shared__ float tw[16][TOPK];
  __shared__ int   ti[16][TOPK];

  const int tid = threadIdx.x;
  const int qt = blockIdx.x, h = blockIdx.y, b = blockIdx.z;
  const int i0 = qt * 16;

  for (int p = tid; p < 16 * DH; p += 256) {
    int qi = p >> 7, d = p & (DH - 1);
    qs[qi][d] = (float)q16[((size_t)(b * T_ + i0 + qi)) * D_ + h * DH + d];
  }
  if (tid < 16 * TOPK) {
    tw[tid >> 3][tid & 7] = -1e30f;
    ti[tid >> 3][tid & 7] = 0;
  }
  __syncthreads();

  for (int c0 = 0; c0 < NMEM; c0 += 256) {
    for (int p = tid; p < 16 * 256; p += 256) {
      int qi = p >> 8, jj = p & 255;
      const _Float16* kp = &kmem[((size_t)(b * NMEM + c0 + jj)) * D_ + h * DH];
      float sum = 0.0f;
#pragma unroll 8
      for (int d = 0; d < DH; ++d) sum += qs[qi][d] * (float)kp[d];
      simc[qi][jj] = sum * SCALE;
    }
    __syncthreads();
    if (tid < 16) {
      for (int jj = 0; jj < 256; ++jj) {
        float s = simc[tid][jj];
        int am = 0;
        float mv = tw[tid][0];
#pragma unroll
        for (int r2 = 1; r2 < TOPK; ++r2)
          if (tw[tid][r2] < mv) { mv = tw[tid][r2]; am = r2; }
        if (s > mv) { tw[tid][am] = s; ti[tid][am] = c0 + jj; }
      }
    }
    __syncthreads();
  }

  if (tid < 16) {
    float mx = tw[tid][0];
#pragma unroll
    for (int r2 = 1; r2 < TOPK; ++r2) mx = fmaxf(mx, tw[tid][r2]);
    float sum = 0.0f;
#pragma unroll
    for (int r2 = 0; r2 < TOPK; ++r2) {
      float e = __expf(tw[tid][r2] - mx);
      tw[tid][r2] = e;
      sum += e;
    }
    float inv = 1.0f / sum;
#pragma unroll
    for (int r2 = 0; r2 < TOPK; ++r2) tw[tid][r2] *= inv;
  }
  __syncthreads();

  for (int p = tid; p < 16 * DH; p += 256) {
    int qi = p >> 7, d = p & (DH - 1);
    float acc = 0.0f;
#pragma unroll
    for (int r2 = 0; r2 < TOPK; ++r2)
      acc += tw[qi][r2] *
             (float)vmem[((size_t)(b * NMEM + ti[qi][r2])) * D_ + h * DH + d];
    cat[((size_t)(b * T_ + i0 + qi)) * (2 * D_) + D_ + h * DH + d] =
        (_Float16)acc;
  }
}

// ------------------------------- launcher -----------------------------------
extern "C" void kernel_launch(void* const* d_in, const int* in_sizes, int n_in,
                              void* d_out, int out_size, void* d_ws,
                              size_t ws_size, hipStream_t stream) {
  using namespace sma;
  (void)in_sizes; (void)n_in; (void)out_size; (void)ws_size;

  const float* hs   = (const float*)d_in[0];
  const float* cosb = (const float*)d_in[1];
  const float* sinb = (const float*)d_in[2];
  const float* Wq   = (const float*)d_in[3];
  const float* Wk   = (const float*)d_in[4];
  const float* Wv   = (const float*)d_in[5];
  const float* Wo   = (const float*)d_in[6];
  const float* Wf   = (const float*)d_in[7];
  const float* bf   = (const float*)d_in[8];
  const float* mem  = (const float*)d_in[9];
  float* out = (float*)d_out;

  // ---- workspace carve-up (~184 MB) ----
  char* p = (char*)d_ws;
  auto take = [&](size_t bytes) { char* q = p; p += bytes; return q; };
  _Float16* hs16    = (_Float16*)take(BTD * 2);
  _Float16* mem16   = (_Float16*)take(BND * 2);
  _Float16* wq16    = (_Float16*)take(DD * 2);
  _Float16* wk16    = (_Float16*)take(DD * 2);
  _Float16* wv16    = (_Float16*)take(DD * 2);
  _Float16* wo16    = (_Float16*)take(DD * 2);
  _Float16* wf16    = (_Float16*)take(D2D * 2);
  float*    Qf      = (float*)take(BTD * 4);
  float*    Kf      = (float*)take(BTD * 4);
  _Float16* q16     = (_Float16*)take(BTD * 2);
  _Float16* k16     = (_Float16*)take(BTD * 2);
  _Float16* v16     = (_Float16*)take(BTD * 2);
  _Float16* kmem16  = (_Float16*)take(BND * 2);
  _Float16* vmem16  = (_Float16*)take(BND * 2);
  _Float16* oloc16  = (_Float16*)take(BTD * 2);
  _Float16* cat16   = (_Float16*)take(BTD * 2 * 2);  // (B,T,2D)

  // ---- f32 -> f16 conversions ----
  sma_cvt_f16<<<2048, 256, 0, stream>>>(hs,  hs16,  BTD);
  sma_cvt_f16<<<2048, 256, 0, stream>>>(mem, mem16, BND);
  sma_cvt_f16<<<2048, 256, 0, stream>>>(Wq,  wq16,  DD);
  sma_cvt_f16<<<2048, 256, 0, stream>>>(Wk,  wk16,  DD);
  sma_cvt_f16<<<2048, 256, 0, stream>>>(Wv,  wv16,  DD);
  sma_cvt_f16<<<2048, 256, 0, stream>>>(Wo,  wo16,  DD);
  sma_cvt_f16<<<2048, 256, 0, stream>>>(Wf,  wf16,  D2D);

  dim3 gBT(D_ / 128, (int)(BT / 128));   // (16,16) — M = B*T rows
  dim3 gBN(D_ / 128, (int)(BN / 128));   // (16,32) — M = B*N rows

  // ---- projections (WMMA) ----
  sma_gemm_nt<<<gBT, 256, 0, stream>>>(hs16, wq16, Qf, nullptr, nullptr, D_, D_);
  sma_gemm_nt<<<gBT, 256, 0, stream>>>(hs16, wk16, Kf, nullptr, nullptr, D_, D_);
  sma_gemm_nt<<<gBT, 256, 0, stream>>>(hs16, wv16, nullptr, v16, nullptr, D_, D_);
  sma_gemm_nt<<<gBN, 256, 0, stream>>>(mem16, wk16, nullptr, kmem16, nullptr, D_, D_);
  sma_gemm_nt<<<gBN, 256, 0, stream>>>(mem16, wv16, nullptr, vmem16, nullptr, D_, D_);

  // ---- RoPE on q,k (fused with f16 cast) ----
  sma_rope_f16<<<2048, 256, 0, stream>>>(Qf, cosb, sinb, q16);
  sma_rope_f16<<<2048, 256, 0, stream>>>(Kf, cosb, sinb, k16);

  // ---- local banded attention ----
  sma_local_attn<<<dim3(T_ / 16, H_, B_), 256, 0, stream>>>(q16, k16, v16, oloc16);

  // ---- o_local @ Wo^T -> cat[:, :D] (f16, ldc = 2D) ----
  sma_gemm_nt<<<gBT, 256, 0, stream>>>(oloc16, wo16, nullptr, cat16, nullptr,
                                       D_, 2 * D_);

  // ---- memory attention -> cat[:, D:2D] ----
  sma_mem_attn<<<dim3(T_ / 16, H_, B_), 256, 0, stream>>>(q16, kmem16, vmem16,
                                                          cat16);

  // ---- final: cat @ Wf^T + bf -> out (f32) ----
  sma_gemm_nt<<<gBT, 256, 0, stream>>>(cat16, wf16, out, nullptr, bf,
                                       2 * D_, D_);
}
